// LIF_Net_48455821033992
// MI455X (gfx1250) — matrix operand order repaired and checked by reading, not compiled, not run on hardware
//
#include <hip/hip_runtime.h>

// ---------------------------------------------------------------------------
// SNN (LIF) forward, fused for MI455X / gfx1250 (wave32, WMMA).
//   kernel 1: h = x @ W1^T + b1   via v_wmma_f32_16x16x4_f32 (exact f32),
//             16x64 strip per wave (4 N-tiles share each A fragment).
//   kernel 2: 100-step LIF scan. 16 waves/block, each owns K=32 of the hidden
//             layer: v1/h state in VGPRs in A-fragment order, W2 resident in
//             one B-fragment register set, 1 WMMA per wave per step,
//             16-way LDS reduction with a double-buffered partial tile so the
//             serial scan needs only ONE barrier per step.
// ---------------------------------------------------------------------------

typedef __attribute__((ext_vector_type(16))) _Float16 v16h;
typedef __attribute__((ext_vector_type(8)))  float    v8f;
typedef __attribute__((ext_vector_type(4)))  float    v4f;
typedef __attribute__((ext_vector_type(2)))  float    v2f;

#define T_STEPS 100
#define BATCH   1024
#define IN_F    784
#define HID     512
#define OUTF    10

// ============================ Kernel 1 =====================================
// h[m][n] = sum_k x[m][k] * W1[n][k] + b1[n];  M=1024, N=512, K=784=196*4.
// One wave per 16(M) x 64(N) strip: 4 accumulators, A fragment reused 4x.
// 512 wave-strips, 4 waves/block -> 128 blocks.
__global__ __launch_bounds__(128) void gemm1_kernel(
    const float* __restrict__ x, const float* __restrict__ W1,
    const float* __restrict__ b1, float* __restrict__ h)
{
  const int lane  = threadIdx.x & 31;
  const int wave  = threadIdx.x >> 5;
  const int tile  = blockIdx.x * 4 + wave;  // 0..511
  const int mtile = tile >> 3;              // 64 M-tiles
  const int ng    = tile & 7;               // 8 N-groups of 64
  const int m0 = mtile * 16, n0 = ng * 64;
  const int rc = lane & 15;                 // A row / B col
  const int hi = lane >> 4;                 // half-wave select
  const int koff = hi * 2;                  // f32 16x16x4 K sub-select

  v8f c0 = {}, c1 = {}, c2 = {}, c3 = {};
  const float* xp  = x  + (size_t)(m0 + rc) * IN_F;
  const float* wp0 = W1 + (size_t)(n0 + rc) * IN_F;
  const float* wp1 = wp0 + (size_t)16 * IN_F;
  const float* wp2 = wp0 + (size_t)32 * IN_F;
  const float* wp3 = wp0 + (size_t)48 * IN_F;

  for (int k = 0; k < IN_F; k += 4) {
    v2f a  = *(const v2f*)(xp  + k + koff);
    v2f b0 = *(const v2f*)(wp0 + k + koff);
    v2f b1v= *(const v2f*)(wp1 + k + koff);
    v2f b2v= *(const v2f*)(wp2 + k + koff);
    v2f b3 = *(const v2f*)(wp3 + k + koff);
    c0 = __builtin_amdgcn_wmma_f32_16x16x4_f32(false, a, false, b0,  (short)0, c0, false, false);
    c1 = __builtin_amdgcn_wmma_f32_16x16x4_f32(false, a, false, b1v, (short)0, c1, false, false);
    c2 = __builtin_amdgcn_wmma_f32_16x16x4_f32(false, a, false, b2v, (short)0, c2, false, false);
    c3 = __builtin_amdgcn_wmma_f32_16x16x4_f32(false, a, false, b3,  (short)0, c3, false, false);
  }

  // C/D layout: VGPR r -> M = r + 8*hi, N = lane&15
#pragma unroll
  for (int r = 0; r < 8; ++r) {
    const size_t mrow = (size_t)(m0 + r + hi * 8) * HID;
    h[mrow + n0 +  0 + rc] = c0[r] + b1[n0 +  0 + rc];
    h[mrow + n0 + 16 + rc] = c1[r] + b1[n0 + 16 + rc];
    h[mrow + n0 + 32 + rc] = c2[r] + b1[n0 + 32 + rc];
    h[mrow + n0 + 48 + rc] = c3[r] + b1[n0 + 48 + rc];
  }
}

// ============================ Kernel 2 =====================================
// Fused 100-step LIF scan. Grid: 64 blocks (16 batch rows each), 512 threads
// (16 waves). Wave w owns hidden K-slice [32w, 32w+32): v1/h for
// (16 rows x 32 k) in 16 VGPRs each, ordered exactly as WMMA A fragments.
// Per step: LIF update -> 1 v_wmma_f32_16x16x32_f16 -> partial to LDS ->
// one barrier -> threads 0..159 reduce 16 partials and run layer-2 LIF.
__global__ __launch_bounds__(512) void lif_scan_kernel(
    const float* __restrict__ h, const float* __restrict__ W2,
    const float* __restrict__ b2, float* __restrict__ out)
{
  const int tid  = threadIdx.x;
  const int lane = tid & 31;
  const int wave = tid >> 5;           // 0..15 : K-slice index
  const int m0   = blockIdx.x * 16;
  const int row  = lane & 15;
  const int col  = lane & 15;
  const int hi   = lane >> 4;
  const int hi8  = hi * 8;
  const int kbase = wave * 32;

  __shared__ float part[2 * 16 * 256]; // double-buffered 16 partial tiles, 32 KB

  // h in A-fragment order: reg[s] <-> K = kbase + (s<8?0:16) + hi8 + (s&7)
  float hreg[16], v1r[16];
  {
    const float* p = h + (size_t)(m0 + row) * HID + kbase;
    v4f g0 = *(const v4f*)(p + hi8 + 0);
    v4f g1 = *(const v4f*)(p + hi8 + 4);
    v4f g2 = *(const v4f*)(p + 16 + hi8 + 0);
    v4f g3 = *(const v4f*)(p + 16 + hi8 + 4);
#pragma unroll
    for (int e = 0; e < 4; ++e) {
      hreg[0  + e] = g0[e];
      hreg[4  + e] = g1[e];
      hreg[8  + e] = g2[e];
      hreg[12 + e] = g3[e];
    }
  }
#pragma unroll
  for (int i = 0; i < 16; ++i) v1r[i] = 0.f;

  // W2 as one B fragment (time-invariant): B[k][n] = W2[n][k], n>=10 -> 0.
  // B 32x16 f16 layout: slot s <-> K = (s<8?0:16) + 8*hi + (s&7), N = lane&15.
  v16h bfrag;
#pragma unroll
  for (int s = 0; s < 16; ++s) {
    int K = kbase + (s < 8 ? 0 : 16) + hi8 + (s & 7);
    bfrag[s] = (col < OUTF) ? (_Float16)W2[col * HID + K] : (_Float16)0.f;
  }

  // Second-layer state: 16 rows x 10 outs = 160 elements, one per thread.
  const int  row2 = tid / OUTF;
  const int  o    = tid % OUTF;
  const int  pidx = row2 * 16 + o;
  const bool red  = (tid < 16 * OUTF);
  const float bb  = red ? b2[o] : 0.f;
  float v2 = 0.f, acc = 0.f;

  for (int t = 0; t < T_STEPS; ++t) {
    float* buf = &part[(t & 1) * 4096];

    // LIF layer 1 on this wave's 16 state registers + pack spikes as f16 A
    v16h af;
#pragma unroll
    for (int s = 0; s < 16; ++s) {
      float v = v1r[s];
      v = v + (hreg[s] - v) * 0.5f;              // leaky integrate (TAU=2)
      float sp = (v >= 1.0f) ? 1.0f : 0.0f;      // fire at V_TH=1
      af[s] = (_Float16)sp;
      v1r[s] = v * (1.0f - sp);                  // hard reset
    }
    v8f C = {};
    C = __builtin_amdgcn_wmma_f32_16x16x32_f16(
            false, af, false, bfrag, (short)0, C, false, false);
#pragma unroll
    for (int r = 0; r < 8; ++r)
      buf[wave * 256 + (r + hi8) * 16 + col] = C[r];

    __syncthreads();   // single barrier per step (double-buffered WAR safety)

    if (red) {
      float y = bb;
#pragma unroll
      for (int w = 0; w < 16; ++w) y += buf[w * 256 + pidx];
      v2 = v2 + (y - v2) * 0.5f;                 // LIF layer 2
      float sp = (v2 >= 1.0f) ? 1.0f : 0.0f;
      acc += sp;
      v2 *= (1.0f - sp);
    }
  }

  if (red) out[(size_t)(m0 + row2) * OUTF + o] = acc * (1.0f / (float)T_STEPS);
}

// ============================ Launcher =====================================
extern "C" void kernel_launch(void* const* d_in, const int* in_sizes, int n_in,
                              void* d_out, int out_size, void* d_ws, size_t ws_size,
                              hipStream_t stream) {
  const float* x  = (const float*)d_in[0];   // [1024,1,28,28] -> [1024,784]
  const float* W1 = (const float*)d_in[1];   // [512,784]
  const float* b1 = (const float*)d_in[2];   // [512]
  const float* W2 = (const float*)d_in[3];   // [10,512]
  const float* b2 = (const float*)d_in[4];   // [10]
  float* out = (float*)d_out;                // [1024,10]
  float* h   = (float*)d_ws;                 // [1024,512] scratch (2 MB)

  // 512 wave-strips of 16x64 / 4 waves per block
  gemm1_kernel<<<128, 128, 0, stream>>>(x, W1, b1, h);
  // 64 blocks x 16 batch rows, 16 WMMA waves each
  lif_scan_kernel<<<BATCH / 16, 512, 0, stream>>>(h, W2, b2, out);
}
